// GotenNet_7464653160727
// MI455X (gfx1250) — matrix operand order repaired and checked by reading, not compiled, not run on hardware
//
#include <hip/hip_runtime.h>
#include <hip/hip_bf16.h>
#include <math.h>

// ---------------- model constants ----------------
#define NODE   128
#define EREF   32
#define NRBF   32
#define HEADS  8
#define DMAX   2
#define NLAYER 2
#define FCUT   5.0f
#define NN     8000
#define NE     64000
#define NG     64
#define SDIM   5            // 1 + 2*DMAX

typedef __attribute__((ext_vector_type(16))) __bf16 v16bf;
typedef __attribute__((ext_vector_type(8)))  __bf16 v8bf;
typedef __attribute__((ext_vector_type(8)))  float  v8f;

__device__ __forceinline__ float silu_f(float x) { return x / (1.0f + __expf(-x)); }

// order-preserving float->uint encode for atomicMax-based segment max
__device__ __forceinline__ unsigned fenc(float f) {
    unsigned u = __float_as_uint(f);
    return (u & 0x80000000u) ? ~u : (u | 0x80000000u);
}
__device__ __forceinline__ float fdec(unsigned u) {
    u = (u & 0x80000000u) ? (u & 0x7FFFFFFFu) : ~u;
    return __uint_as_float(u);
}

__device__ __forceinline__ v8bf pack8(float4 a, float4 b, bool inb) {
    v8bf p;
    p[0] = (__bf16)(inb ? a.x : 0.0f);
    p[1] = (__bf16)(inb ? a.y : 0.0f);
    p[2] = (__bf16)(inb ? a.z : 0.0f);
    p[3] = (__bf16)(inb ? a.w : 0.0f);
    p[4] = (__bf16)(inb ? b.x : 0.0f);
    p[5] = (__bf16)(inb ? b.y : 0.0f);
    p[6] = (__bf16)(inb ? b.z : 0.0f);
    p[7] = (__bf16)(inb ? b.w : 0.0f);
    return p;
}

// ================= WMMA GEMM =================
// C[M,Ncol] = act(A[M,K] @ B[K,Ncol] + bias)
// 256 threads = 8 wave32; block tile 128x64; K-step 32; bf16 staged in LDS.
// Wave (wm in 0..3, wn in 0..1) owns a 32x32 output tile -> 4 accumulators,
// 4 v_wmma per K-step from 2 A-fragments x 2 B-fragments (2x matrix work per
// barrier vs a 16x32 wave tile). K must be a multiple of 32 (true here:
// 32/64/128/256). Staging is branch-free: addresses are clamped in-range and
// OOB lanes zeroed with selects -> straight global_load_b128/ds_store_b128.
#define TSM 128
#define TSN 64
#define TSK 32

__global__ __launch_bounds__(256)
void k_gemm(const float* __restrict__ A, const float* __restrict__ B,
            const float* __restrict__ bias, float* __restrict__ C,
            int M, int Ncol, int K, int act)
{
    __shared__ __attribute__((aligned(32))) __bf16 lA[TSM * TSK]; // [m][k] 8KB
    __shared__ __attribute__((aligned(32))) __bf16 lB[TSN * TSK]; // [n][k] 4KB

    const int tid  = threadIdx.x;
    const int lane = tid & 31;
    const int wave = tid >> 5;
    const int wm = wave >> 1, wn = wave & 1;
    const int g  = lane >> 4;       // half-wave group
    const int r  = lane & 15;
    const int m0 = blockIdx.x * TSM;
    const int n0 = blockIdx.y * TSN;

    // A staging: one thread = 16 consecutive K of one row (2 threads/row)
    const int arow = tid >> 1, aks = (tid & 1) * 16;
    const int agr  = m0 + arow;
    const bool ainb = agr < M;
    const float* asrc = A + (size_t)(ainb ? agr : (M - 1)) * K + aks;
    // B staging: one thread = one n column, 8 consecutive K (coalesced loads)
    const int bn = tid & 63, bkb = (tid >> 6) * 8;
    const int bgn = n0 + bn;
    const bool binb = bgn < Ncol;
    const float* bsrc = B + (size_t)bkb * Ncol + (binb ? bgn : (Ncol - 1));

    v8f acc00 = {}, acc01 = {}, acc10 = {}, acc11 = {};

    for (int kt = 0; kt < K; kt += TSK) {
        {   // stage A tile (4 b128 loads -> pack -> 2 b128 LDS stores)
            const float4 a0 = *(const float4*)(asrc + kt);
            const float4 a1 = *(const float4*)(asrc + kt + 4);
            const float4 a2 = *(const float4*)(asrc + kt + 8);
            const float4 a3 = *(const float4*)(asrc + kt + 12);
            *(v8bf*)&lA[arow * TSK + aks]     = pack8(a0, a1, ainb);
            *(v8bf*)&lA[arow * TSK + aks + 8] = pack8(a2, a3, ainb);
            if (kt + TSK < K)
                __builtin_prefetch(asrc + kt + TSK, 0, 0);
        }
        {   // stage B tile transposed (8 coalesced b32 loads -> one b128 store)
            const float* s = bsrc + (size_t)kt * Ncol;
            v8bf p;
            #pragma unroll
            for (int u = 0; u < 8; ++u) {
                float x = s[(size_t)u * Ncol];
                p[u] = (__bf16)(binb ? x : 0.0f);
            }
            *(v8bf*)&lB[bn * TSK + bkb] = p;
        }
        __syncthreads();

        // A fragments (ISA 16-bit A 16x32 layout): halves 0..7 -> K=g*8+0..7,
        // halves 8..15 -> K=16+g*8+0..7, row = lane%16 (+16 for 2nd fragment)
        union { v16bf v; v8bf h[2]; } af0, af1;
        af0.h[0] = *(const v8bf*)&lA[(wm * 32 + r) * TSK + g * 8];
        af0.h[1] = *(const v8bf*)&lA[(wm * 32 + r) * TSK + 16 + g * 8];
        af1.h[0] = *(const v8bf*)&lA[(wm * 32 + 16 + r) * TSK + g * 8];
        af1.h[1] = *(const v8bf*)&lA[(wm * 32 + 16 + r) * TSK + 16 + g * 8];
        // B fragments: halves 0..7 -> K=g*8+0..7, halves 8..15 -> K=16+g*8+0..7, col = lane%16 (+16 for 2nd)
        const v16bf b0 = __builtin_shufflevector(*(const v8bf*)&lB[(wn * 32 + r) * TSK + g * 8], *(const v8bf*)&lB[(wn * 32 + r) * TSK + 16 + g * 8], 0,1,2,3,4,5,6,7,8,9,10,11,12,13,14,15);
        const v16bf b1 = __builtin_shufflevector(*(const v8bf*)&lB[(wn * 32 + 16 + r) * TSK + g * 8], *(const v8bf*)&lB[(wn * 32 + 16 + r) * TSK + 16 + g * 8], 0,1,2,3,4,5,6,7,8,9,10,11,12,13,14,15);

        acc00 = __builtin_amdgcn_wmma_f32_16x16x32_bf16(false, af0.v, false, b0,
                                                        (short)0, acc00, false, false);
        acc01 = __builtin_amdgcn_wmma_f32_16x16x32_bf16(false, af0.v, false, b1,
                                                        (short)0, acc01, false, false);
        acc10 = __builtin_amdgcn_wmma_f32_16x16x32_bf16(false, af1.v, false, b0,
                                                        (short)0, acc10, false, false);
        acc11 = __builtin_amdgcn_wmma_f32_16x16x32_bf16(false, af1.v, false, b1,
                                                        (short)0, acc11, false, false);
        __syncthreads();
    }

    // C layout: VGPR j -> M = g*8 + j, N = lane%16 (coalesced b32 stores)
    #pragma unroll
    for (int mi = 0; mi < 2; ++mi) {
        #pragma unroll
        for (int j = 0; j < 8; ++j) {
            const int m = m0 + wm * 32 + mi * 16 + g * 8 + j;
            if (m >= M) continue;
            #pragma unroll
            for (int nt = 0; nt < 2; ++nt) {
                const int n = n0 + wn * 32 + nt * 16 + r;
                if (n >= Ncol) continue;
                float v;
                if (mi == 0) v = nt ? acc01[j] : acc00[j];
                else         v = nt ? acc11[j] : acc10[j];
                if (bias) v += bias[n];
                if (act)  v = silu_f(v);
                C[(size_t)m * Ncol + n] = v;
            }
        }
    }
}

// ================= elementwise / segment kernels =================

// y[n,:] = W[idx[n],:] + b  (one-hot @ W collapsed to a row gather)
__global__ void k_embed(const int* __restrict__ idx, const float* __restrict__ W,
                        const float* __restrict__ b, float* __restrict__ y, int M)
{
    int i = blockIdx.x * 256 + threadIdx.x;
    if (i >= M * NODE) return;
    int n = i >> 7, c = i & 127;
    y[i] = W[idx[n] * NODE + c] + b[c];
}

// per-edge geometry: rl[e,0..7] = [r1(3), sph2(5)], cc[e], rbf[e,0..31]
__global__ void k_geom(const float* __restrict__ xe, const float* __restrict__ centers,
                       const float* __restrict__ gamma, float* __restrict__ rl,
                       float* __restrict__ cc, float* __restrict__ rbf)
{
    int e = blockIdx.x * 256 + threadIdx.x;
    if (e >= NE) return;
    float x = xe[e * 3 + 0], y = xe[e * 3 + 1], z = xe[e * 3 + 2];
    float r0 = sqrtf(x * x + y * y + z * z);
    float inv = 1.0f / r0;
    x *= inv; y *= inv; z *= inv;
    const float s3 = 1.7320508075688772f;
    rl[e * 8 + 0] = x;
    rl[e * 8 + 1] = y;
    rl[e * 8 + 2] = z;
    rl[e * 8 + 3] = s3 * x * z;
    rl[e * 8 + 4] = s3 * x * y;
    rl[e * 8 + 5] = y * y - 0.5f * (x * x + z * z);
    rl[e * 8 + 6] = s3 * y * z;
    rl[e * 8 + 7] = 0.5f * s3 * (z * z - x * x);
    cc[e] = (r0 <= FCUT) ? 0.5f * (cosf(3.14159265358979f * r0 / FCUT) + 1.0f) : 0.0f;
    #pragma unroll
    for (int j = 0; j < NRBF; ++j) {
        float d = r0 - centers[j];
        rbf[e * NRBF + j] = __expf(-gamma[j] * d * d);
    }
}

// m_i[ni[e],c] += hN[nj[e],c] * proj[e,c] * cc[e]
__global__ void k_msg_scatter(const float* __restrict__ hN, const float* __restrict__ proj,
                              const float* __restrict__ cc, const int* __restrict__ nj,
                              const int* __restrict__ ni, float* __restrict__ m_i)
{
    int i = blockIdx.x * 256 + threadIdx.x;
    if (i >= NE * NODE) return;
    int e = i >> 7, c = i & 127;
    atomicAdd(&m_i[(size_t)ni[e] * NODE + c],
              hN[(size_t)nj[e] * NODE + c] * proj[i] * cc[e]);
}

__global__ void k_concat2(const float* __restrict__ a, const float* __restrict__ b,
                          float* __restrict__ y, int M, int da, int db)
{
    int i = blockIdx.x * 256 + threadIdx.x;
    int w = da + db;
    if (i >= M * w) return;
    int row = i / w, c = i - row * w;
    y[i] = (c < da) ? a[(size_t)row * da + c] : b[(size_t)row * db + (c - da)];
}

// one wave32 per 128-wide row; optional trailing SiLU
__global__ void k_layernorm(const float* __restrict__ x, const float* __restrict__ gw,
                            const float* __restrict__ bw, float* __restrict__ y,
                            int M, int act)
{
    int wave = threadIdx.x >> 5, lane = threadIdx.x & 31;
    int row = blockIdx.x * 8 + wave;
    if (row >= M) return;
    const float* xr = x + (size_t)row * NODE;
    float v[4]; float s = 0.0f;
    #pragma unroll
    for (int u = 0; u < 4; ++u) { v[u] = xr[lane + u * 32]; s += v[u]; }
    #pragma unroll
    for (int o = 16; o; o >>= 1) s += __shfl_xor(s, o, 32);
    float mean = s * (1.0f / NODE);
    float q = 0.0f;
    #pragma unroll
    for (int u = 0; u < 4; ++u) { float d = v[u] - mean; q += d * d; }
    #pragma unroll
    for (int o = 16; o; o >>= 1) q += __shfl_xor(q, o, 32);
    float is = rsqrtf(q * (1.0f / NODE) + 1e-5f);
    float* yr = y + (size_t)row * NODE;
    #pragma unroll
    for (int u = 0; u < 4; ++u) {
        int c = lane + u * 32;
        float r = (v[u] - mean) * is * gw[c] + bw[c];
        yr[c] = act ? silu_f(r) : r;
    }
}

// t[e,c] = (h[ni]+h[nj]) * erp[e,c]
__global__ void k_edge_t(const float* __restrict__ h, const float* __restrict__ erp,
                         const int* __restrict__ nj, const int* __restrict__ ni,
                         float* __restrict__ t)
{
    int i = blockIdx.x * 256 + threadIdx.x;
    if (i >= NE * NODE) return;
    int e = i >> 7, c = i & 127;
    t[i] = (h[(size_t)ni[e] * NODE + c] + h[(size_t)nj[e] * NODE + c]) * erp[i];
}

// attention logits per (edge, head) + segment max
__global__ void k_logits_max(const float* __restrict__ q, const float* __restrict__ k,
                             const float* __restrict__ re, const int* __restrict__ nj,
                             const int* __restrict__ ni, float* __restrict__ logits,
                             unsigned* __restrict__ smax)
{
    int i = blockIdx.x * 256 + threadIdx.x;
    if (i >= NE * HEADS) return;
    int e = i >> 3, hd = i & 7;
    const float* qr = q + (size_t)ni[e] * NODE + hd * 16;
    const float* kr = k + (size_t)nj[e] * NODE + hd * 16;
    const float* rr = re + (size_t)e * NODE + hd * 16;
    float a = 0.0f;
    #pragma unroll
    for (int d = 0; d < 16; ++d) a += qr[d] * kr[d] * rr[d];
    logits[i] = a;
    atomicMax(&smax[ni[e] * HEADS + hd], fenc(a));
}

__global__ void k_expsum(float* __restrict__ logits, const unsigned* __restrict__ smax,
                         float* __restrict__ ssum, const int* __restrict__ ni)
{
    int i = blockIdx.x * 256 + threadIdx.x;
    if (i >= NE * HEADS) return;
    int e = i >> 3, hd = i & 7;
    float m = fdec(smax[ni[e] * HEADS + hd]);
    float ex = __expf(logits[i] - m);
    logits[i] = ex;
    atomicAdd(&ssum[ni[e] * HEADS + hd], ex);
}

// o[e,c] = attn(e,head(c)) * v[nj,c] + o[e,c] * ms[nj,c] * cc[e]   (tw pre-stored in o)
__global__ void k_sea_combine(float* __restrict__ o, const float* __restrict__ logits,
                              const float* __restrict__ ssum, const float* __restrict__ v,
                              const float* __restrict__ ms, const float* __restrict__ cc,
                              const int* __restrict__ nj, const int* __restrict__ ni,
                              int outdim)
{
    size_t i = (size_t)blockIdx.x * 256 + threadIdx.x;
    if (i >= (size_t)NE * outdim) return;
    int e = (int)(i / outdim), c = (int)(i - (size_t)e * outdim);
    int dv = outdim >> 3;
    int hd = c / dv;
    float attn = logits[e * HEADS + hd] / (ssum[ni[e] * HEADS + hd] + 1e-16f);
    o[i] = attn * v[(size_t)nj[e] * outdim + c] + o[i] * ms[(size_t)nj[e] * outdim + c] * cc[e];
}

__global__ void k_scatter_h(float* __restrict__ h, const float* __restrict__ o,
                            const int* __restrict__ ni, int outdim)
{
    int i = blockIdx.x * 256 + threadIdx.x;
    if (i >= NE * NODE) return;
    int e = i >> 7, c = i & 127;
    atomicAdd(&h[(size_t)ni[e] * NODE + c], o[(size_t)e * outdim + c]);
}

// embedding: X[i][ni,d,:] += o[:, i*128 + c] * rl[e,d]
__global__ void k_x_init(float* __restrict__ X0, float* __restrict__ X1,
                         const float* __restrict__ o, const float* __restrict__ rl,
                         const int* __restrict__ ni)
{
    size_t i = (size_t)blockIdx.x * 256 + threadIdx.x;
    if (i >= (size_t)NE * 8 * NODE) return;
    int e = (int)(i >> 10);
    int d = (int)((i >> 7) & 7);
    int c = (int)(i & 127);
    int ii = (d < 3) ? 0 : 1;
    int dl = ii ? d - 3 : d;
    float val = o[(size_t)e * (2 * NODE) + ii * NODE + c] * rl[e * 8 + d];
    float* X = ii ? &X1[((size_t)ni[e] * 5 + dl) * NODE + c]
                  : &X0[((size_t)ni[e] * 3 + dl) * NODE + c];
    atomicAdd(X, val);
}

// layer update: X += seg( od*rl + ot*Xprev[nj] )
__global__ void k_x_upd(float* __restrict__ X0, float* __restrict__ X1,
                        const float* __restrict__ Xp0, const float* __restrict__ Xp1,
                        const float* __restrict__ o, const float* __restrict__ rl,
                        const int* __restrict__ nj, const int* __restrict__ ni)
{
    size_t i = (size_t)blockIdx.x * 256 + threadIdx.x;
    if (i >= (size_t)NE * 8 * NODE) return;
    int e = (int)(i >> 10);
    int d = (int)((i >> 7) & 7);
    int c = (int)(i & 127);
    int ii = (d < 3) ? 0 : 1;
    int dl = ii ? d - 3 : d;
    int dim = ii ? 5 : 3;
    const float* Xp = ii ? Xp1 : Xp0;
    float od = o[(size_t)e * (SDIM * NODE) + (1 + ii) * NODE + c];
    float ot = o[(size_t)e * (SDIM * NODE) + (1 + DMAX + ii) * NODE + c];
    float val = od * rl[e * 8 + d] + ot * Xp[((size_t)nj[e] * dim + dl) * NODE + c];
    float* X = ii ? &X1[((size_t)ni[e] * 5 + dl) * NODE + c]
                  : &X0[((size_t)ni[e] * 3 + dl) * NODE + c];
    atomicAdd(X, val);
}

// s[e,c] = sum_d eq[ni,d,c] * ek[nj,d,c]   (d over 3+5 degrees)
__global__ void k_tlogits(const float* __restrict__ eq0, const float* __restrict__ eq1,
                          const float* __restrict__ ek0, const float* __restrict__ ek1,
                          const int* __restrict__ nj, const int* __restrict__ ni,
                          float* __restrict__ s)
{
    int i = blockIdx.x * 256 + threadIdx.x;
    if (i >= NE * EREF) return;
    int e = i >> 5, c = i & 31;
    int a = ni[e], b = nj[e];
    float acc = 0.0f;
    #pragma unroll
    for (int d = 0; d < 3; ++d)
        acc += eq0[((size_t)a * 3 + d) * EREF + c] * ek0[((size_t)b * 3 + d) * EREF + c];
    #pragma unroll
    for (int d = 0; d < 5; ++d)
        acc += eq1[((size_t)a * 5 + d) * EREF + c] * ek1[((size_t)b * 5 + d) * EREF + c];
    s[i] = acc;
}

__global__ void k_t_fma(float* __restrict__ t, const float* __restrict__ a,
                        const float* __restrict__ b)
{
    int i = blockIdx.x * 256 + threadIdx.x;
    if (i >= NE * NODE) return;
    t[i] += a[i] * b[i];
}

__global__ void k_nrm(const float* __restrict__ Xvu0, const float* __restrict__ Xvu1,
                      float* __restrict__ nrm)
{
    int i = blockIdx.x * 256 + threadIdx.x;
    if (i >= NN * NODE) return;
    int n = i >> 7, c = i & 127;
    float s = 0.0f;
    #pragma unroll
    for (int d = 0; d < 3; ++d) { float v = Xvu0[((size_t)n * 3 + d) * NODE + c]; s += v * v; }
    #pragma unroll
    for (int d = 0; d < 5; ++d) { float v = Xvu1[((size_t)n * 5 + d) * NODE + c]; s += v * v; }
    nrm[i] = sqrtf(s) + 1e-8f;
}

__global__ void k_h_add(float* __restrict__ h, const float* __restrict__ m)
{
    int i = blockIdx.x * 256 + threadIdx.x;
    if (i >= NN * NODE) return;
    h[i] += m[(size_t)(i >> 7) * (2 * NODE) + (i & 127)];
}

__global__ void k_x_eqff(float* __restrict__ X0, float* __restrict__ X1,
                         const float* __restrict__ Xvu0, const float* __restrict__ Xvu1,
                         const float* __restrict__ m)
{
    size_t i = (size_t)blockIdx.x * 256 + threadIdx.x;
    if (i >= (size_t)NN * 8 * NODE) return;
    int n = (int)(i >> 10);
    int d = (int)((i >> 7) & 7);
    int c = (int)(i & 127);
    int ii = (d < 3) ? 0 : 1;
    int dl = ii ? d - 3 : d;
    int dim = ii ? 5 : 3;
    float m2 = m[(size_t)n * (2 * NODE) + NODE + c];
    size_t idx = ((size_t)n * dim + dl) * NODE + c;
    if (ii) X1[idx] += m2 * Xvu1[idx];
    else    X0[idx] += m2 * Xvu0[idx];
}

__global__ void k_pool(float* __restrict__ out, const float* __restrict__ yv,
                       const int* __restrict__ batch)
{
    int i = blockIdx.x * 256 + threadIdx.x;
    if (i >= NN) return;
    atomicAdd(&out[batch[i]], yv[i]);
}

// ================= host-side wiring =================

struct MLPp { const float *b1, *b2, *w1, *w2; };
struct SeaP { const float *ln_b, *ln_g; MLPp mv; const float *wk, *wq, *wre; };
struct LayerP {
    MLPp mlp_m; const float* w_vu;                       // eqff
    MLPp mlp_t, mlp_w; const float *wvk0, *wvk1, *wvq;   // gata.htr
    MLPp mlp_s; SeaP sea; const float* w_rs;             // gata
};

extern "C" void kernel_launch(void* const* d_in, const int* in_sizes, int n_in,
                              void* d_out, int out_size, void* d_ws, size_t ws_size,
                              hipStream_t stream)
{
    (void)in_sizes; (void)n_in; (void)out_size; (void)ws_size;

    // ---- params in JAX tree-flatten order (dicts sorted by key) ----
    int pi = 0;
    auto F = [&]() { return (const float*)d_in[pi++]; };
    auto I = [&]() { return (const int*)d_in[pi++]; };
    auto mlpP = [&]() { MLPp m; m.b1 = F(); m.b2 = F(); m.w1 = F(); m.w2 = F(); return m; };
    auto seaP = [&]() { SeaP s; s.ln_b = F(); s.ln_g = F(); s.mv = mlpP();
                        s.wk = F(); s.wq = F(); s.wre = F(); return s; };

    MLPp dec = mlpP();                                   // params.dec
    // params.emb (alphabetical)
    const float* a_na_b  = F(); const float* a_na_w  = F();
    const float* a_nbr_b = F(); const float* a_nbr_w = F();
    const float* centers = F(); const float* gamma   = F();
    const float* e_ln_b  = F(); const float* e_ln_g  = F();
    MLPp e_mlp_s = mlpP();
    SeaP e_sea   = seaP();
    const float* w_erp = F(); const float* w_ndp = F();
    const float* w_nrd = F(); const float* w_nru = F();
    const float* e_w_rs = F();
    // params.layers
    LayerP lay[NLAYER];
    for (int l = 0; l < NLAYER; ++l) {
        lay[l].mlp_m = mlpP(); lay[l].w_vu = F();        // eqff
        lay[l].mlp_t = mlpP(); lay[l].mlp_w = mlpP();    // gata.htr
        lay[l].wvk0 = F(); lay[l].wvk1 = F(); lay[l].wvq = F();
        lay[l].mlp_s = mlpP(); lay[l].sea = seaP(); lay[l].w_rs = F();
    }
    const float* xe = F();
    const int* xn = I();
    const int* eidx = I();
    const int* batch = I();
    const int* nj = eidx;
    const int* ni = eidx + NE;

    // ---- workspace bump allocator ----
    char* wsb = (char*)d_ws;
    size_t off = 0;
    auto alloc = [&](size_t elems) {
        float* p = (float*)(wsb + off);
        off += ((elems * sizeof(float)) + 255) & ~(size_t)255;
        return p;
    };
    float* hA   = alloc((size_t)NN * NODE);
    float* hN   = alloc((size_t)NN * NODE);
    float* rl   = alloc((size_t)NE * 8);
    float* cc   = alloc(NE);
    float* rbf  = alloc((size_t)NE * NRBF);
    float* proj = alloc((size_t)NE * NODE);   // reused as edge-MLP hidden
    float* mw   = alloc((size_t)NE * NODE);
    float* mt   = alloc((size_t)NE * NODE);
    float* t    = alloc((size_t)NE * NODE);
    float* re   = alloc((size_t)NE * NODE);
    float* o    = alloc((size_t)NE * SDIM * NODE);
    float* logits = alloc((size_t)NE * HEADS);
    unsigned* smax = (unsigned*)alloc((size_t)NN * HEADS);
    float* ssum = alloc((size_t)NN * HEADS);
    float* m_i  = alloc((size_t)NN * NODE);
    float* cat  = alloc((size_t)NN * 2 * NODE);
    float* pre  = alloc((size_t)NN * NODE);
    float* h    = alloc((size_t)NN * NODE);
    float* hn   = alloc((size_t)NN * NODE);
    float* qb   = alloc((size_t)NN * NODE);
    float* kb   = alloc((size_t)NN * NODE);
    float* vmid = alloc((size_t)NN * NODE);
    float* v    = alloc((size_t)NN * SDIM * NODE);
    float* msmid= alloc((size_t)NN * NODE);
    float* ms   = alloc((size_t)NN * SDIM * NODE);
    float* X0   = alloc((size_t)NN * 3 * NODE);
    float* X1   = alloc((size_t)NN * 5 * NODE);
    float* Xp0  = alloc((size_t)NN * 3 * NODE);
    float* Xp1  = alloc((size_t)NN * 5 * NODE);
    float* Xvu0 = alloc((size_t)NN * 3 * NODE);
    float* Xvu1 = alloc((size_t)NN * 5 * NODE);
    float* eq0  = alloc((size_t)NN * 3 * EREF);
    float* eq1  = alloc((size_t)NN * 5 * EREF);
    float* ek0  = alloc((size_t)NN * 3 * EREF);
    float* ek1  = alloc((size_t)NN * 5 * EREF);
    float* s32  = alloc((size_t)NE * EREF);
    float* nrm  = alloc((size_t)NN * NODE);
    float* mmout= alloc((size_t)NN * 2 * NODE);
    float* yv   = alloc(NN);

    auto gemm = [&](const float* A, const float* B, const float* bias, float* C,
                    int M, int Ncol, int K, int act) {
        dim3 g((M + TSM - 1) / TSM, (Ncol + TSN - 1) / TSN);
        k_gemm<<<g, 256, 0, stream>>>(A, B, bias, C, M, Ncol, K, act);
    };
    auto g1 = [](size_t n) { return dim3((unsigned)((n + 255) / 256)); };

    // ---------- embedding block ----------
    k_embed<<<g1((size_t)NN * NODE), 256, 0, stream>>>(xn, a_na_w,  a_na_b,  hA, NN);
    k_embed<<<g1((size_t)NN * NODE), 256, 0, stream>>>(xn, a_nbr_w, a_nbr_b, hN, NN);
    k_geom<<<g1(NE), 256, 0, stream>>>(xe, centers, gamma, rl, cc, rbf);
    gemm(rbf, w_ndp, nullptr, proj, NE, NODE, NRBF, 0);
    hipMemsetAsync(m_i, 0, (size_t)NN * NODE * 4, stream);
    k_msg_scatter<<<g1((size_t)NE * NODE), 256, 0, stream>>>(hN, proj, cc, nj, ni, m_i);
    k_concat2<<<g1((size_t)NN * 2 * NODE), 256, 0, stream>>>(hA, m_i, cat, NN, NODE, NODE);
    gemm(cat, w_nrd, nullptr, pre, NN, NODE, 2 * NODE, 0);
    k_layernorm<<<(NN + 7) / 8, 256, 0, stream>>>(pre, e_ln_g, e_ln_b, hn, NN, 1);
    gemm(hn, w_nru, nullptr, h, NN, NODE, NODE, 0);
    gemm(rbf, w_erp, nullptr, re, NE, NODE, NRBF, 0);
    k_edge_t<<<g1((size_t)NE * NODE), 256, 0, stream>>>(h, re, nj, ni, t);

    auto run_sea = [&](const SeaP& sp, const MLPp& msp, const float* w_rs, int outdim) {
        k_layernorm<<<(NN + 7) / 8, 256, 0, stream>>>(h, sp.ln_g, sp.ln_b, hn, NN, 0);
        gemm(hn, sp.wq, nullptr, qb, NN, NODE, NODE, 0);
        gemm(hn, sp.wk, nullptr, kb, NN, NODE, NODE, 0);
        gemm(hn, sp.mv.w1, sp.mv.b1, vmid, NN, NODE, NODE, 1);
        gemm(vmid, sp.mv.w2, sp.mv.b2, v, NN, outdim, NODE, 0);
        gemm(t, sp.wre, nullptr, re, NE, NODE, NODE, 1);
        hipMemsetAsync(smax, 0, (size_t)NN * HEADS * 4, stream);
        hipMemsetAsync(ssum, 0, (size_t)NN * HEADS * 4, stream);
        k_logits_max<<<g1((size_t)NE * HEADS), 256, 0, stream>>>(qb, kb, re, nj, ni, logits, smax);
        k_expsum<<<g1((size_t)NE * HEADS), 256, 0, stream>>>(logits, smax, ssum, ni);
        gemm(t, w_rs, nullptr, o, NE, outdim, NODE, 0);          // tw -> o (fused in-place)
        gemm(h, msp.w1, msp.b1, msmid, NN, NODE, NODE, 1);
        gemm(msmid, msp.w2, msp.b2, ms, NN, outdim, NODE, 0);
        k_sea_combine<<<g1((size_t)NE * outdim), 256, 0, stream>>>(o, logits, ssum, v, ms,
                                                                   cc, nj, ni, outdim);
    };

    run_sea(e_sea, e_mlp_s, e_w_rs, DMAX * NODE);
    hipMemsetAsync(X0, 0, (size_t)NN * 3 * NODE * 4, stream);
    hipMemsetAsync(X1, 0, (size_t)NN * 5 * NODE * 4, stream);
    k_x_init<<<g1((size_t)NE * 8 * NODE), 256, 0, stream>>>(X0, X1, o, rl, ni);

    // ---------- layers ----------
    for (int l = 0; l < NLAYER; ++l) {
        const LayerP& L = lay[l];
        gemm(X0, L.wvq,  nullptr, eq0, NN * 3, EREF, NODE, 0);
        gemm(X1, L.wvq,  nullptr, eq1, NN * 5, EREF, NODE, 0);
        gemm(X0, L.wvk0, nullptr, ek0, NN * 3, EREF, NODE, 0);
        gemm(X1, L.wvk1, nullptr, ek1, NN * 5, EREF, NODE, 0);
        k_tlogits<<<g1((size_t)NE * EREF), 256, 0, stream>>>(eq0, eq1, ek0, ek1, nj, ni, s32);
        gemm(s32, L.mlp_w.w1, L.mlp_w.b1, proj, NE, NODE, EREF, 1);
        gemm(proj, L.mlp_w.w2, L.mlp_w.b2, mw, NE, NODE, NODE, 0);
        gemm(t, L.mlp_t.w1, L.mlp_t.b1, proj, NE, NODE, NODE, 1);
        gemm(proj, L.mlp_t.w2, L.mlp_t.b2, mt, NE, NODE, NODE, 0);
        k_t_fma<<<g1((size_t)NE * NODE), 256, 0, stream>>>(t, mw, mt);

        run_sea(L.sea, L.mlp_s, L.w_rs, SDIM * NODE);
        k_scatter_h<<<g1((size_t)NE * NODE), 256, 0, stream>>>(h, o, ni, SDIM * NODE);
        hipMemcpyAsync(Xp0, X0, (size_t)NN * 3 * NODE * 4, hipMemcpyDeviceToDevice, stream);
        hipMemcpyAsync(Xp1, X1, (size_t)NN * 5 * NODE * 4, hipMemcpyDeviceToDevice, stream);
        k_x_upd<<<g1((size_t)NE * 8 * NODE), 256, 0, stream>>>(X0, X1, Xp0, Xp1, o, rl, nj, ni);

        // eqff
        gemm(X0, L.w_vu, nullptr, Xvu0, NN * 3, NODE, NODE, 0);
        gemm(X1, L.w_vu, nullptr, Xvu1, NN * 5, NODE, NODE, 0);
        k_nrm<<<g1((size_t)NN * NODE), 256, 0, stream>>>(Xvu0, Xvu1, nrm);
        k_concat2<<<g1((size_t)NN * 2 * NODE), 256, 0, stream>>>(nrm, h, cat, NN, NODE, NODE);
        gemm(cat, L.mlp_m.w1, L.mlp_m.b1, msmid, NN, NODE, 2 * NODE, 1);
        gemm(msmid, L.mlp_m.w2, L.mlp_m.b2, mmout, NN, 2 * NODE, NODE, 0);
        k_h_add<<<g1((size_t)NN * NODE), 256, 0, stream>>>(h, mmout);
        k_x_eqff<<<g1((size_t)NN * 8 * NODE), 256, 0, stream>>>(X0, X1, Xvu0, Xvu1, mmout);
    }

    // ---------- decoder + graph pooling ----------
    gemm(h, dec.w1, dec.b1, vmid, NN, NODE, NODE, 1);
    gemm(vmid, dec.w2, dec.b2, yv, NN, 1, NODE, 0);
    hipMemsetAsync(d_out, 0, NG * 4, stream);
    k_pool<<<g1(NN), 256, 0, stream>>>((float*)d_out, yv, batch);
}
